// STDPDigitCaps_69106023792956
// MI455X (gfx1250) — compile-verified, matrix operand order checked by reading, not run on hardware
//
#include <hip/hip_runtime.h>

// ---------------------------------------------------------------------------
// Problem sizes (from reference): B=512 T=64 I=32 D=16 O=32 H=32
//   GEMM: M=T=64, K=I*D=512, N=O*H=1024, per batch element b (512 blocks)
// ---------------------------------------------------------------------------
#define NB   512
#define NT   64
#define NI   32
#define ND   16
#define NO   32
#define NH   32
#define NK   512   // I*D
#define NN   1024  // O*H

#define A_PRE   0.60653065971f  // exp(-1/2)
#define A_POST  0.60653065971f
#define ETA     0.05f

// per-wave staging scratch: 64 rows x 20 floats (16 data + 4 pad for banks/align)
#define SCR_STRIDE 20
#define SCR_WAVE   (NT * SCR_STRIDE)   // 1280 floats = 5120 B per wave

typedef __bf16 bf16_t;
typedef __attribute__((ext_vector_type(8)))  bf16_t v8bf_t;
typedef __attribute__((ext_vector_type(16))) bf16_t v16bf_t;
typedef __attribute__((ext_vector_type(8)))  float  v8f_t;

static __device__ __forceinline__ v16bf_t load_frag(const bf16_t* p) {
  v8bf_t lo = *(const v8bf_t*)(p);
  v8bf_t hi = *(const v8bf_t*)(p + 16);
  return __builtin_shufflevector(lo, hi, 0,1,2,3,4,5,6,7,8,9,10,11,12,13,14,15);
}

// ---------------------------------------------------------------------------
// K0: repack W[O][I][H][D] (f32) -> Wt[n=o*32+h][k=i*16+d] (bf16, K-contiguous)
// ---------------------------------------------------------------------------
__global__ __launch_bounds__(256) void convw_kernel(const float* __restrict__ W,
                                                    unsigned short* __restrict__ Wt_raw) {
  bf16_t* Wt = (bf16_t*)Wt_raw;
  int idx = blockIdx.x * 256 + threadIdx.x;          // n*512 + k, total 524288
  int n = idx >> 9, k = idx & (NK - 1);
  int o = n >> 5, h = n & 31, i = k >> 4, d = k & 15;
  Wt[idx] = (bf16_t)W[((o * NI + i) * NH + h) * ND + d];
}

// ---------------------------------------------------------------------------
// per-ntile epilogue: column sums -> usum_g (direct), stage 64x16 u_hat block
// into per-wave LDS scratch (conflict-free, per-wave DS in-order), then row
// sum-of-squares for the two lane-owned rows (t=lane, t=lane+32).
// ---------------------------------------------------------------------------
static __device__ __forceinline__ void tile_epilogue(const v8f_t acc[4], int coln,
                                                     int b, int lane, int lrow, int khalf,
                                                     float* sw, float* __restrict__ usum_g,
                                                     float& sq_t0, float& sq_t1) {
  float cs = 0.0f;
  #pragma unroll
  for (int mt = 0; mt < 4; ++mt)
    #pragma unroll
    for (int r = 0; r < 8; ++r) cs += acc[mt][r];
  cs += __shfl_xor(cs, 16, 32);                    // combine the two row-halves
  if (lane < 16) usum_g[(size_t)b * NN + coln] = cs;

  #pragma unroll
  for (int mt = 0; mt < 4; ++mt)
    #pragma unroll
    for (int r = 0; r < 8; ++r)
      sw[(mt * 16 + khalf + r) * SCR_STRIDE + lrow] = acc[mt][r];

  const float4* r0 = (const float4*)(sw + lane * SCR_STRIDE);
  const float4* r1 = (const float4*)(sw + (lane + 32) * SCR_STRIDE);
  float s0 = 0.f, s1 = 0.f;
  #pragma unroll
  for (int q = 0; q < 4; ++q) {
    float4 a = r0[q], c = r1[q];
    s0 += a.x * a.x + a.y * a.y + a.z * a.z + a.w * a.w;
    s1 += c.x * c.x + c.y * c.y + c.z * c.z + c.w * c.w;
  }
  sq_t0 += s0;
  sq_t1 += s1;
}

// ---------------------------------------------------------------------------
// K1: per-b GEMM with v_wmma_f32_16x16x32_bf16 + fused, atomic-free reductions.
// ntiles processed in PAIRS so each A fragment (LDS) feeds two WMMAs -> halves
// LDS read traffic in the hot loop. Loop pragmas prevent the full-unroll +
// cross-iteration CSE that spilled A fragments to scratch in the last round.
// Dynamic LDS: A(64x512 bf16)=64KB | per-wave scratch 8x5120B=40KB -> 104KB
// ---------------------------------------------------------------------------
__global__ __launch_bounds__(256) void gemm_kernel(const float* __restrict__ spikes,
                                                   const unsigned short* __restrict__ Wt_raw,
                                                   float* __restrict__ usum_g,
                                                   float* __restrict__ post_g,
                                                   float* __restrict__ pre_g) {
  extern __shared__ char smem[];
  bf16_t* As      = (bf16_t*)smem;                  // [64][512]
  float*  scratch = (float*)(smem + NT * NK * 2);   // [8 waves][64][20]
  const bf16_t* Wt = (const bf16_t*)Wt_raw;

  const int b   = blockIdx.x;
  const int tid = threadIdx.x;

  // ---- load spikes[b] (64x512 f32), convert to bf16 in LDS, emit pre ----
  const float* sp = spikes + (size_t)b * NT * NK;
  #pragma unroll
  for (int c = 0; c < 8; ++c) {
    int cell = tid + c * 256;                   // (t,i) cell: t=cell>>5, i=cell&31
    const float4* src = (const float4*)(sp + cell * ND);
    bf16_t* dst = As + cell * ND;
    float s = 0.0f;
    #pragma unroll
    for (int q = 0; q < 4; ++q) {
      float4 v = src[q];
      s += v.x + v.y + v.z + v.w;
      dst[q * 4 + 0] = (bf16_t)v.x;
      dst[q * 4 + 1] = (bf16_t)v.y;
      dst[q * 4 + 2] = (bf16_t)v.z;
      dst[q * 4 + 3] = (bf16_t)v.w;
    }
    pre_g[(size_t)b * (NT * NI) + cell] = s;
  }
  __syncthreads();

  const int wave  = tid >> 5;
  const int lane  = tid & 31;
  const int lrow  = lane & 15;         // A row / B col / C col within tile
  const int khalf = (lane >> 4) * 8;   // K sub-offset per 16-bit WMMA layout
  float* sw = scratch + wave * SCR_WAVE;   // this wave's private staging tile

  // register Σu² accumulators: lane owns rows t=lane and t=lane+32,
  // for the wave's 4 output-group columns o = wave*4 + jp
  float sq0[4] = {0.f, 0.f, 0.f, 0.f};
  float sq1[4] = {0.f, 0.f, 0.f, 0.f};

  #pragma clang loop unroll(disable)
  for (int jp = 0; jp < 4; ++jp) {
    const int col0 = (wave * 8 + jp * 2) * 16 + lrow;   // even ntile's column
    const bf16_t* B0 = Wt + (size_t)col0 * NK;
    const bf16_t* B1 = B0 + (size_t)16 * NK;            // odd ntile's column

    v8f_t acc0[4], acc1[4];
    #pragma unroll
    for (int mt = 0; mt < 4; ++mt) {
      acc0[mt] = (v8f_t){0,0,0,0,0,0,0,0};
      acc1[mt] = (v8f_t){0,0,0,0,0,0,0,0};
    }

    #pragma clang loop unroll_count(2)
    for (int kb = 0; kb < NK / 32; ++kb) {
      const int k0 = kb * 32 + khalf;
      v16bf_t bf0 = load_frag(B0 + k0);
      v16bf_t bf1 = load_frag(B1 + k0);
      #pragma unroll
      for (int mt = 0; mt < 4; ++mt) {
        v16bf_t af = load_frag(As + (mt * 16 + lrow) * NK + k0);  // shared by pair
        acc0[mt] = __builtin_amdgcn_wmma_f32_16x16x32_bf16(
            false, af, false, bf0, (short)0, acc0[mt], false, false);
        acc1[mt] = __builtin_amdgcn_wmma_f32_16x16x32_bf16(
            false, af, false, bf1, (short)0, acc1[mt], false, false);
      }
    }

    tile_epilogue(acc0, col0,      b, lane, lrow, khalf, sw, usum_g, sq0[jp], sq1[jp]);
    tile_epilogue(acc1, col0 + 16, b, lane, lrow, khalf, sw, usum_g, sq0[jp], sq1[jp]);
  }

  // ---- emit post[b][t][o] = (||u|| >= 0.5)  <=>  (Σu² >= 0.25) ----
  #pragma unroll
  for (int p = 0; p < 4; ++p) {
    const int o = wave * 4 + p;
    post_g[(size_t)b * (NT * NO) + lane * NO + o]        = (sq0[p] >= 0.25f) ? 1.0f : 0.0f;
    post_g[(size_t)b * (NT * NO) + (lane + 32) * NO + o] = (sq1[p] >= 0.25f) ? 1.0f : 0.0f;
  }
}

// ---------------------------------------------------------------------------
// K2: STDP scan + softmax + column sum -> coeff[b][o]. One block per b,
//     1024 threads: thread = i*32 + o, so each wave32 is one softmax row.
// ---------------------------------------------------------------------------
__global__ __launch_bounds__(1024) void stdp_kernel(const float* __restrict__ pre_g,
                                                    const float* __restrict__ post_g,
                                                    float* __restrict__ coeff_g) {
  __shared__ float pre_s[NT * NI], post_s[NT * NO];
  __shared__ float ptr_s[NT * NI], potr_s[NT * NO];
  __shared__ float coeff_s[NO];
  const int b = blockIdx.x, tid = threadIdx.x;

  for (int idx = tid; idx < NT * NI; idx += 1024) {
    pre_s[idx]  = pre_g [(size_t)b * (NT * NI) + idx];
    post_s[idx] = post_g[(size_t)b * (NT * NO) + idx];
  }
  if (tid < NO) coeff_s[tid] = 0.0f;
  __syncthreads();

  if (tid < NI) {                       // pre traces
    float r = 0.0f;
    for (int t = 0; t < NT; ++t) { r = A_PRE * r + pre_s[t * NI + tid]; ptr_s[t * NI + tid] = r; }
  } else if (tid < NI + NO) {           // post traces
    int o = tid - NI; float r = 0.0f;
    for (int t = 0; t < NT; ++t) { r = A_POST * r + post_s[t * NO + o]; potr_s[t * NO + o] = r; }
  }
  __syncthreads();

  const int i = tid >> 5, o = tid & 31;
  float acc = 0.0f;
  for (int t = 0; t < NT; ++t)
    acc += ptr_s[t * NI + i] * post_s[t * NO + o] + potr_s[t * NO + o] * pre_s[t * NI + i];
  float bv = ETA * acc;

  // wave32 softmax over o (lanes of this wave == row i)
  float mx = bv;
  #pragma unroll
  for (int d = 16; d > 0; d >>= 1) mx = fmaxf(mx, __shfl_xor(mx, d, 32));
  float e = __expf(bv - mx);
  float se = e;
  #pragma unroll
  for (int d = 16; d > 0; d >>= 1) se += __shfl_xor(se, d, 32);
  float c = e / se;

  atomicAdd(&coeff_s[o], c);            // coeff[b,o] = sum_i softmax(b,i,:)[o]
  __syncthreads();
  if (tid < NO) coeff_g[(size_t)b * NO + tid] = coeff_s[tid];
}

// ---------------------------------------------------------------------------
// K3: v[b,o,h] = coeff[b,o] * usum[b, o*32+h] / T
// ---------------------------------------------------------------------------
__global__ __launch_bounds__(256) void final_kernel(const float* __restrict__ usum,
                                                    const float* __restrict__ coeff,
                                                    float* __restrict__ out) {
  int idx = blockIdx.x * 256 + threadIdx.x;   // 512*1024 = 524288
  int b = idx >> 10, n = idx & (NN - 1), o = n >> 5;
  out[idx] = coeff[b * NO + o] * usum[idx] * (1.0f / (float)NT);
}

// ---------------------------------------------------------------------------
// Workspace layout (bytes):
//   Wt   bf16 [1024][512]        @ 0          (1,048,576)
//   usum f32  [512][1024]        @ 1,048,576  (2,097,152)
//   pre  f32  [512][64][32]      @ 3,145,728  (4,194,304)
//   post f32  [512][64][32]      @ 7,340,032  (4,194,304)
//   coeff f32 [512][32]          @ 11,534,336 (65,536)     total ~11.6 MB
// ---------------------------------------------------------------------------
extern "C" void kernel_launch(void* const* d_in, const int* in_sizes, int n_in,
                              void* d_out, int out_size, void* d_ws, size_t ws_size,
                              hipStream_t stream) {
  const float* spikes = (const float*)d_in[0];   // [512,64,32,16] f32
  const float* W      = (const float*)d_in[1];   // [32,32,32,16]  f32
  float* out = (float*)d_out;                    // [512,32,32]    f32
  char* ws = (char*)d_ws;

  unsigned short* Wt = (unsigned short*)(ws);
  float* usum  = (float*)(ws + 1048576);
  float* pre   = (float*)(ws + 3145728);
  float* post  = (float*)(ws + 7340032);
  float* coeff = (float*)(ws + 11534336);

  convw_kernel<<<2048, 256, 0, stream>>>(W, Wt);

  const size_t gemm_lds = (size_t)NT * NK * 2 + (size_t)8 * SCR_WAVE * 4; // 106,496 B
  gemm_kernel<<<NB, 256, gemm_lds, stream>>>(spikes, Wt, usum, post, pre);

  stdp_kernel<<<NB, 1024, 0, stream>>>(pre, post, coeff);

  final_kernel<<<2048, 256, 0, stream>>>(usum, coeff, out);
}
